// ChemicallyInformedLoss_35854386987433
// MI455X (gfx1250) — compile-verified, matrix operand order checked by reading, not run on hardware
//
#include <hip/hip_runtime.h>
#include <hip/hip_bf16.h>

// ---------------------------------------------------------------------------
// Problem constants (from reference setup_inputs)
// ---------------------------------------------------------------------------
#define N_ROWS 8192
#define M_COLS 128
#define F_DIM  1024
#define SIM_TAU 0.8f
#define EPS_F  1e-8f

typedef __attribute__((ext_vector_type(16))) __bf16 v16bf;
typedef __attribute__((ext_vector_type(8)))  __bf16 v8bf;
typedef __attribute__((ext_vector_type(8)))  float  v8f;

// ---------------------------------------------------------------------------
// Helpers
// ---------------------------------------------------------------------------
__device__ __forceinline__ float block_reduce_sum(float v, float* sred) {
    const int t = threadIdx.x;
    sred[t] = v;
    __syncthreads();
    for (int s = blockDim.x >> 1; s > 0; s >>= 1) {
        if (t < s) sred[t] += sred[t + s];
        __syncthreads();
    }
    float r = sred[0];
    __syncthreads();
    return r;
}

// Load a 16x32 bf16 WMMA operand fragment from a row-major matrix.
// ISA 16-bit A layout: lanes 0-15 hold K {0..7,16..23} of row (lane),
// lanes 16-31 hold K {8..15,24..31} of row (lane-16). B of X*Y^T loaded
// symmetrically from rows of Y.
__device__ __forceinline__ v16bf load_frag(const __bf16* __restrict__ base,
                                           int rowBase, int ld, int k0, int lane) {
    const int r   = lane & 15;
    const int sel = lane >> 4;  // 0 or 1
    const __bf16* p = base + (size_t)(rowBase + r) * (size_t)ld + k0 + (sel << 3);
    v8bf lo = *(const v8bf*)(p);        // K offsets sel*8   .. sel*8+7
    v8bf hi = *(const v8bf*)(p + 16);   // K offsets sel*8+16.. sel*8+23
    return __builtin_shufflevector(lo, hi, 0,1,2,3,4,5,6,7,8,9,10,11,12,13,14,15);
}

__device__ __forceinline__ v8f wmma_bf16(v16bf a, v16bf b, v8f c) {
    return __builtin_amdgcn_wmma_f32_16x16x32_bf16(
        /*neg_a=*/false, a, /*neg_b=*/false, b,
        /*c_mod=*/(short)0, c, /*reuse_a=*/false, /*reuse_b=*/false);
}

// ---------------------------------------------------------------------------
// Kernel 0: zero the scalar accumulators and column sums
// ---------------------------------------------------------------------------
__global__ void cil_zero(float* acc, float* colp, float* colt) {
    const int t = threadIdx.x;
    if (t < 16) acc[t] = 0.0f;
    if (t < M_COLS) { colp[t] = 0.0f; colt[t] = 0.0f; }
}

// ---------------------------------------------------------------------------
// Kernel 1: per-row prep. One block (256 threads) per row i.
//  - y_pred = sigmoid(logits); store bf16 row-major + transposed, y_true transposed
//  - weighted BCE partial -> acc[0]
//  - sq[i] = sum y_pred^2 ; Lsample term -> acc[1]
//  - column sums of y_pred / y_true via atomics
//  - f_norm = features/(||f||+eps) stored bf16
// ---------------------------------------------------------------------------
__global__ __launch_bounds__(256)
void cil_prep(const float* __restrict__ logits, const float* __restrict__ y_true,
              const float* __restrict__ features, const float* __restrict__ cw,
              float* colp, float* colt, float* sq, float* acc,
              __bf16* __restrict__ ypb, __bf16* __restrict__ ypT,
              __bf16* __restrict__ ytT, __bf16* __restrict__ fnb) {
    __shared__ float sred[256];
    const int i = blockIdx.x;
    const int t = threadIdx.x;

    float bcew = 0.0f, sqv = 0.0f, ep = 0.0f, lc = 0.0f;
    if (t < M_COLS) {
        const float lg = logits[(size_t)i * M_COLS + t];
        const float yt = y_true[(size_t)i * M_COLS + t];
        const float yp = 1.0f / (1.0f + expf(-lg));
        ypb[(size_t)i * M_COLS + t]  = (__bf16)yp;
        ypT[(size_t)t * N_ROWS + i]  = (__bf16)yp;
        ytT[(size_t)t * N_ROWS + i]  = (__bf16)yt;
        const float bce = fmaxf(lg, 0.0f) - lg * yt + log1pf(expf(-fabsf(lg)));
        bcew = cw[t] * bce;
        sqv  = yp * yp;
        ep   = yp;
        lc   = yt;
        atomicAdd(&colp[t], yp);
        atomicAdd(&colt[t], yt);
    }

    float s;
    s = block_reduce_sum(bcew, sred);
    if (t == 0) atomicAdd(&acc[0], s);
    s = block_reduce_sum(sqv, sred);
    if (t == 0) sq[i] = s;
    const float epsum = block_reduce_sum(ep, sred);
    const float lcsum = block_reduce_sum(lc, sred);
    if (t == 0) {
        // E1 + E2*label_counts - E_pred, relu, squared
        float d = fmaxf((1.0f + 1.0f * lcsum) - epsum, 0.0f);
        atomicAdd(&acc[1], d * d);
    }

    // Feature normalization: 1024 elems, 4 per thread (coalesced stride 256)
    float fv[4];
    float ss = 0.0f;
#pragma unroll
    for (int j = 0; j < 4; ++j) {
        fv[j] = features[(size_t)i * F_DIM + t + 256 * j];
        ss += fv[j] * fv[j];
    }
    s = block_reduce_sum(ss, sred);
    const float inv = 1.0f / (sqrtf(s) + EPS_F);
#pragma unroll
    for (int j = 0; j < 4; ++j)
        fnb[(size_t)i * F_DIM + t + 256 * j] = (__bf16)(fv[j] * inv);
}

// ---------------------------------------------------------------------------
// Kernel 2: fused N x N pass. Each wave computes a 32x32 output tile
// (2 A-frags x 2 B-frags -> 4 WMMA accumulators) of
// sim = f_norm @ f_norm^T (K=1024) and gram = y_pred @ y_pred^T (K=128)
// via v_wmma_f32_16x16x32_bf16, then reduces masked dist2 in-register.
// Register-level reuse: 4 fragment loads feed 4 WMMAs (2 b128/lane per WMMA).
// Block = 8 waves arranged 4x2 -> 128x64 tile; grid = (64, 128).
// ---------------------------------------------------------------------------
__global__ __launch_bounds__(256)
void cil_pair(const __bf16* __restrict__ fnb, const __bf16* __restrict__ ypb,
              const float* __restrict__ sq, float* acc) {
    __shared__ float sred[256];
    const int lane = threadIdx.x & 31;
    const int wid  = threadIdx.x >> 5;
    const int rowBase = blockIdx.x * 128 + (wid >> 1) * 32;
    const int colBase = blockIdx.y * 64  + (wid & 1)  * 32;

    v8f s00 = {}, s01 = {}, s10 = {}, s11 = {};
#pragma unroll 2
    for (int k = 0; k < F_DIM; k += 32) {
        v16bf a0 = load_frag(fnb, rowBase,      F_DIM, k, lane);
        v16bf a1 = load_frag(fnb, rowBase + 16, F_DIM, k, lane);
        v16bf b0 = load_frag(fnb, colBase,      F_DIM, k, lane);
        v16bf b1 = load_frag(fnb, colBase + 16, F_DIM, k, lane);
        s00 = wmma_bf16(a0, b0, s00);
        s01 = wmma_bf16(a0, b1, s01);
        s10 = wmma_bf16(a1, b0, s10);
        s11 = wmma_bf16(a1, b1, s11);
    }

    v8f g00 = {}, g01 = {}, g10 = {}, g11 = {};
#pragma unroll
    for (int k = 0; k < M_COLS; k += 32) {
        v16bf a0 = load_frag(ypb, rowBase,      M_COLS, k, lane);
        v16bf a1 = load_frag(ypb, rowBase + 16, M_COLS, k, lane);
        v16bf b0 = load_frag(ypb, colBase,      M_COLS, k, lane);
        v16bf b1 = load_frag(ypb, colBase + 16, M_COLS, k, lane);
        g00 = wmma_bf16(a0, b0, g00);
        g01 = wmma_bf16(a0, b1, g01);
        g10 = wmma_bf16(a1, b0, g10);
        g11 = wmma_bf16(a1, b1, g11);
    }

    // C/D layout: lane element r -> (M = r + 8*(lane>>4), N = lane&15)
    const int sel = lane >> 4;
    const int n   = lane & 15;
    const float sqj0 = sq[colBase + n];
    const float sqj1 = sq[colBase + 16 + n];
    float sumD = 0.0f, cnt = 0.0f;
#pragma unroll
    for (int r = 0; r < 8; ++r) {
        const int i0 = rowBase + r + 8 * sel;
        const int i1 = i0 + 16;
        const float sqi0 = sq[i0];
        const float sqi1 = sq[i1];
        if (s00[r] > SIM_TAU) { sumD += sqi0 + sqj0 - 2.0f * g00[r]; cnt += 1.0f; }
        if (s01[r] > SIM_TAU) { sumD += sqi0 + sqj1 - 2.0f * g01[r]; cnt += 1.0f; }
        if (s10[r] > SIM_TAU) { sumD += sqi1 + sqj0 - 2.0f * g10[r]; cnt += 1.0f; }
        if (s11[r] > SIM_TAU) { sumD += sqi1 + sqj1 - 2.0f * g11[r]; cnt += 1.0f; }
    }
    const float S = block_reduce_sum(sumD, sred);
    const float C = block_reduce_sum(cnt, sred);
    if (threadIdx.x == 0) {
        atomicAdd(&acc[2], S);
        atomicAdd(&acc[3], C);
    }
}

// ---------------------------------------------------------------------------
// Kernel 3: Lcol. corr_pred/corr_true are 128x128 = y^T y / N, computed from
// the transposed bf16 copies (rows = original columns, K = 8192). Two WMMA
// accumulators per wave, fused squared-difference reduction.
// Grid = (2, 4) blocks of 8 waves (64x32 tiles).
// ---------------------------------------------------------------------------
__global__ __launch_bounds__(256)
void cil_lcol(const __bf16* __restrict__ ypT, const __bf16* __restrict__ ytT,
              float* acc) {
    __shared__ float sred[256];
    const int lane = threadIdx.x & 31;
    const int wid  = threadIdx.x >> 5;
    const int rowBase = blockIdx.x * 64 + (wid >> 1) * 16;
    const int colBase = blockIdx.y * 32 + (wid & 1) * 16;

    v8f cp = {};
    v8f ct = {};
#pragma unroll 2
    for (int k = 0; k < N_ROWS; k += 32) {
        v16bf ap = load_frag(ypT, rowBase, N_ROWS, k, lane);
        v16bf bp = load_frag(ypT, colBase, N_ROWS, k, lane);
        cp = wmma_bf16(ap, bp, cp);
        v16bf at = load_frag(ytT, rowBase, N_ROWS, k, lane);
        v16bf bt = load_frag(ytT, colBase, N_ROWS, k, lane);
        ct = wmma_bf16(at, bt, ct);
    }

    const float invN = 1.0f / (float)N_ROWS;
    float local = 0.0f;
#pragma unroll
    for (int r = 0; r < 8; ++r) {
        const float d = (cp[r] - ct[r]) * invN;
        local += d * d;
    }
    const float S = block_reduce_sum(local, sred);
    if (threadIdx.x == 0) atomicAdd(&acc[4], S);
}

// ---------------------------------------------------------------------------
// Kernel 4: Lclass from column sums + final combination. 1 block, 128 threads.
// ---------------------------------------------------------------------------
__global__ void cil_final(const float* acc, const float* colp, const float* colt,
                          float* out) {
    __shared__ float sred[M_COLS];
    const int t = threadIdx.x;
    const float Nf = (float)N_ROWS;

    const float csp = colp[t];
    const float cst = colt[t];
    const float Ej   = csp / Nf;
    const float bpos = cst;
    const float bneg = Nf - cst;
    // y_true binary: co_diag_pos = batch_pos/N, co_diag_neg = batch_neg/N
    const float minT  = 1.0f + 0.2f * (bpos / Nf);
    const float moutT = 0.2f * (bneg / Nf);
    float pt = fmaxf(Ej - minT, 0.0f);  pt *= pt;
    float nt = fmaxf(moutT - Ej, 0.0f); nt *= nt;
    sred[t] = bpos * pt + bneg * nt;
    __syncthreads();
    for (int s = M_COLS >> 1; s > 0; s >>= 1) {
        if (t < s) sred[t] += sred[t + s];
        __syncthreads();
    }
    if (t == 0) {
        const float Lclass  = sred[0] / Nf;
        const float Lbasis  = acc[0] / (Nf * (float)M_COLS);
        const float Lstt    = (acc[3] > 0.0f) ? acc[2] / (Nf * Nf) : 0.0f;
        const float Lsample = acc[1] / Nf;
        const float Lcol    = acc[4] / ((float)M_COLS * (float)M_COLS);
        const float Ltotal  = Lbasis + 0.3f * Lstt + 0.3f * Lclass
                            + 0.5f * Lsample + 0.3f * Lcol;
        out[0] = Ltotal;  out[1] = Lbasis; out[2] = Lstt;
        out[3] = Lclass;  out[4] = Lsample; out[5] = Lcol;
    }
}

// ---------------------------------------------------------------------------
// Launch
// ---------------------------------------------------------------------------
extern "C" void kernel_launch(void* const* d_in, const int* in_sizes, int n_in,
                              void* d_out, int out_size, void* d_ws, size_t ws_size,
                              hipStream_t stream) {
    (void)in_sizes; (void)n_in; (void)out_size; (void)ws_size;
    const float* logits   = (const float*)d_in[0];
    const float* y_true   = (const float*)d_in[1];
    const float* features = (const float*)d_in[2];
    const float* cw       = (const float*)d_in[3];
    float* out = (float*)d_out;

    // Workspace layout (256B-aligned sections)
    char* w = (char*)d_ws;
    float* acc  = (float*)(w);                       // 16 floats
    float* colp = (float*)(w + 256);                 // 128 floats
    float* colt = (float*)(w + 1024);                // 128 floats
    float* sq   = (float*)(w + 2048);                // 8192 floats
    size_t off = 40960;
    __bf16* ypb = (__bf16*)(w + off);  off += (size_t)N_ROWS * M_COLS * 2;  // 2 MB
    __bf16* ypT = (__bf16*)(w + off);  off += (size_t)M_COLS * N_ROWS * 2;  // 2 MB
    __bf16* ytT = (__bf16*)(w + off);  off += (size_t)M_COLS * N_ROWS * 2;  // 2 MB
    __bf16* fnb = (__bf16*)(w + off);  // 8192*1024*2 = 16 MB

    cil_zero<<<1, 128, 0, stream>>>(acc, colp, colt);
    cil_prep<<<N_ROWS, 256, 0, stream>>>(logits, y_true, features, cw,
                                         colp, colt, sq, acc, ypb, ypT, ytT, fnb);
    cil_pair<<<dim3(N_ROWS / 128, N_ROWS / 64), 256, 0, stream>>>(fnb, ypb, sq, acc);
    cil_lcol<<<dim3(M_COLS / 64, M_COLS / 32), 256, 0, stream>>>(ypT, ytT, acc);
    cil_final<<<1, M_COLS, 0, stream>>>(acc, colp, colt, out);
}
